// ParallelPhysicalRecurrentLayer_17867063952064
// MI455X (gfx1250) — matrix operand order repaired and checked by reading, not compiled, not run on hardware
//
#include <hip/hip_runtime.h>
#include <hip/hip_bf16.h>
#include <math.h>

typedef _Float16 h16;
typedef __attribute__((ext_vector_type(16))) _Float16 v16h;
typedef __attribute__((ext_vector_type(8)))  _Float16 v8h;
typedef __attribute__((ext_vector_type(8)))  float    v8f;

#define LDIM 16
#define CDIM 32
#define HDIM 64
#define WDIM 64
#define WF   33
#define RDIM 16
#define NPIX (HDIM*WDIM)
#define PI_F 3.14159265358979f

// ---------------------------------------------------------------- TDM support
#if __has_builtin(__builtin_amdgcn_tensor_load_to_lds) && \
    __has_builtin(__builtin_amdgcn_s_wait_tensorcnt)
#define HAVE_TDM 1
typedef __attribute__((ext_vector_type(4))) unsigned int v4u;
typedef __attribute__((ext_vector_type(8))) int          v8i;
typedef __attribute__((ext_vector_type(4))) int          v4i;

// 1-D contiguous TDM copy global->LDS, 'bytes' must be a multiple of 8.
// D# per CDNA5 ISA 8.3/8.4: group0 {count=1, lds_addr, global_addr, type=2},
// group1 {data_size=8B, tensor_dim0=tile_dim0=stride0=N, dim1=1}.
__device__ __forceinline__ void tdm_load_1d(const void* gptr, void* lptr,
                                            unsigned bytes) {
  unsigned long long ga = (unsigned long long)(size_t)gptr;
  unsigned ga_lo = __builtin_amdgcn_readfirstlane((unsigned)ga);
  unsigned ga_hi = __builtin_amdgcn_readfirstlane((unsigned)(ga >> 32));
  unsigned lds   = __builtin_amdgcn_readfirstlane((unsigned)(size_t)lptr);
  unsigned n     = bytes >> 3;                     // 8-byte units
  v4u g0;
  g0.x = 1u;                                       // count=1 (user descriptor)
  g0.y = lds;                                      // lds_addr
  g0.z = ga_lo;                                    // global_addr[31:0]
  g0.w = (ga_hi & 0x01FFFFFFu) | (2u << 30);       // global_addr[56:32] | type=2
  v8i g1;
  g1[0] = (int)(3u << 16);                         // data_size = 8B
  g1[1] = (int)((n & 0xFFFFu) << 16);              // tensor_dim0[15:0]
  g1[2] = (int)((n >> 16) | (1u << 16));           // tensor_dim0[31:16] | tensor_dim1[15:0]=1
  g1[3] = (int)(n << 16);                          // tensor_dim1[31:16]=0 | tile_dim0
  g1[4] = 1;                                       // tile_dim1=1, tile_dim2=0
  g1[5] = (int)n;                                  // tensor_dim0_stride[31:0]
  g1[6] = 0; g1[7] = 0;
  v4i z = {0,0,0,0};
#if __clang_major__ >= 23
  v8i z8 = {0,0,0,0,0,0,0,0};
  __builtin_amdgcn_tensor_load_to_lds(g0, g1, z, z, z8, 0);
#else
  __builtin_amdgcn_tensor_load_to_lds(g0, g1, z, z, 0);
#endif
}
__device__ __forceinline__ void tdm_wait() {
  __builtin_amdgcn_s_wait_tensorcnt((short)0);
}
#else
#define HAVE_TDM 0
#endif

// ---------------------------------------------------------------- helpers
__device__ __forceinline__ v8f wmma32(v16h a, v16h b, v8f c) {
  // D = A(16x32 f16) * B(32x16 f16) + C(16x16 f32)
  return __builtin_amdgcn_wmma_f32_16x16x32_f16(false, a, false, b, (short)0, c,
                                                false, false);
}

// A-fragment (16x32 f16). lane<16 holds row m=lane&15,
// K = kb..kb+15, kb=(lane>>4)*16 (contiguous). 'p' = row base + kb.
__device__ __forceinline__ v16h fragA(const h16* p) {
  union { v16h v; v8h h[2]; } u;
  u.h[0] = *(const v8h*)(p);
  u.h[1] = *(const v8h*)(p + 8);
  return u.v;
}

// B-fragment (32x16 f16) from transposed storage Bt[n][k] (ld=32):
// lane holds col n=lane&15, K = kb..kb+15, kb=(lane>>4)*16 (contiguous).
__device__ __forceinline__ v16h fragBt(const h16* bt) {
  union { v16h v; v8h h[2]; } u;
  u.h[0] = *(const v8h*)(bt);
  u.h[1] = *(const v8h*)(bt + 8);
  return u.v;
}

__device__ __forceinline__ float siluf(float x) { return x / (1.f + __expf(-x)); }
__device__ __forceinline__ float sigmf(float x) { return 1.f / (1.f + __expf(-x)); }

// ---------------------------------------------------------------- conv (WMMA)
// 3x3 SAME conv, C=32 -> C=32, + bias + SiLU.  Output layout (l,h,w,c) f16.
template<bool IN_X>
__global__ void k_conv(const float* __restrict__ xin, const h16* __restrict__ fin,
                       const float* __restrict__ wgt, const float* __restrict__ bias,
                       h16* __restrict__ fout)
{
  alignas(16) __shared__ h16 tile[10*68*CDIM];  // [r 0..9][wc 0..67][c]
  alignas(16) __shared__ h16 wt[9*CDIM*CDIM];   // [tap][cout][cin]  (Bt form)
  __shared__ float bsh[CDIM];
  const int l  = blockIdx.x >> 3;
  const int h0 = (blockIdx.x & 7) * 8;
  const int tid = threadIdx.x;

  for (int i = tid; i < 9*CDIM*CDIM; i += 256) {
    int tap = i >> 10; int o = (i >> 5) & 31; int ci = i & 31;
    wt[i] = (h16)wgt[o*(CDIM*9) + ci*9 + tap];       // (O,I,3,3)
  }
  if (tid < CDIM) bsh[tid] = bias[tid];
  for (int i = tid; i < 10*66*CDIM; i += 256) {
    int c, r, wc;
    if (IN_X) { wc = i % 66; int t2 = i / 66; r = t2 % 10; c = t2 / 10; }
    else      { c = i & 31; int t2 = i >> 5; wc = t2 % 66; r = t2 / 66; }
    int gh = h0 + r - 1, gw = wc - 1;
    float v = 0.f;
    if (gh >= 0 && gh < HDIM && gw >= 0 && gw < WDIM) {
      if (IN_X) v = xin[((size_t)(c*LDIM + l)*HDIM + gh)*WDIM + gw];       // (C,L,H,W)
      else      v = (float)fin[(((size_t)(l*HDIM + gh)*WDIM + gw) << 5) + c];
    }
    tile[(r*68 + wc)*CDIM + c] = (h16)v;
  }
  __syncthreads();

  const int lane = tid & 31, wave = tid >> 5;
  const int r = wave;                                  // output row in band
  const int m = lane & 15, kb8 = (lane >> 4) * 16;
  const int colB = lane & 15, kb16 = (lane >> 4) * 16;
  const int ro = (lane >> 4) * 8;
  const int h = h0 + r;

  for (int w0 = 0; w0 < WDIM; w0 += 16) {
    v8f acc0 = {0,0,0,0,0,0,0,0}, acc1 = {0,0,0,0,0,0,0,0};
    #pragma unroll
    for (int tap = 0; tap < 9; ++tap) {
      int dy = tap / 3, dx = tap % 3;
      v16h A  = fragA(&tile[((r + dy)*68 + (w0 + m + dx))*CDIM + kb8]);
      const h16* bt = &wt[tap*CDIM*CDIM];
      v16h B0 = fragBt(&bt[colB*CDIM + kb16]);
      v16h B1 = fragBt(&bt[(16 + colB)*CDIM + kb16]);
      acc0 = wmma32(A, B0, acc0);
      acc1 = wmma32(A, B1, acc1);
    }
    #pragma unroll
    for (int j = 0; j < 8; ++j) {                      // lane holds col=cout, rows j(+8)
      int w = w0 + ro + j;
      size_t base = ((size_t)((l*HDIM + h)*WDIM + w) << 5);
      fout[base + colB]      = (h16)siluf(acc0[j] + bsh[colB]);
      fout[base + 16 + colB] = (h16)siluf(acc1[j] + bsh[16 + colB]);
    }
  }
}

// ---------------------------------------------------------------- glob gate
__global__ void k_glob(const h16* __restrict__ f2, const float* __restrict__ gfc_w,
                       const float* __restrict__ gfc_b, float* __restrict__ glob)
{
  __shared__ float red[256];
  int l = blockIdx.x, t = threadIdx.x;
  int c = t & 31, s = t >> 5;
  float acc = 0.f;
  for (int p = s; p < NPIX; p += 8)
    acc += (float)f2[((size_t)(l*NPIX + p) << 5) + c];
  red[t] = acc;
  __syncthreads();
  if (t < 32) {
    float m = 0.f;
    for (int s2 = 0; s2 < 8; ++s2) m += red[s2*32 + t];
    red[t] = m * (1.f / NPIX);
  }
  __syncthreads();
  if (t < 32) {
    float a = gfc_b[t];
    for (int k = 0; k < 32; ++k) a += red[k] * gfc_w[t*32 + k];
    glob[l*32 + t] = sigmf(a);
  }
}

// ---------------------------------------------------------------- pool -> feat
__global__ void k_pool(const h16* __restrict__ f2, const float* __restrict__ glob,
                       h16* __restrict__ feat)
{
  __shared__ float fm[WDIM][CDIM];
  int l = blockIdx.x, t = threadIdx.x;
  int c = t & 31;
  for (int wb = (t >> 5); wb < WDIM; wb += 8) {
    float s = 0.f;
    for (int h = 0; h < HDIM; ++h)
      s += (float)f2[((size_t)((l*HDIM + h)*WDIM + wb) << 5) + c];
    fm[wb][c] = s * (1.f / HDIM);
  }
  __syncthreads();
  float gl = glob[l*32 + c];
  for (int j = (t >> 5); j < WF; j += 8) {
    int s0 = (j * WDIM) / WF;
    int e0 = ((j + 1) * WDIM + WF - 1) / WF;
    float s = 0.f;
    for (int w2 = s0; w2 < e0; ++w2) s += fm[w2][c];
    feat[(l*WF + j)*CDIM + c] = (h16)(s / (float)(e0 - s0) * gl);
  }
}

// ---------------------------------------------------------------- head GEMM (WMMA)
__global__ void k_head(const h16* __restrict__ feat, const float* __restrict__ head_w,
                       const float* __restrict__ head_b, float* __restrict__ p)
{
  alignas(16) __shared__ h16 at[16*CDIM];
  int mt = blockIdx.x, t = threadIdx.x;
  for (int i = t; i < 16*CDIM; i += 256) at[i] = feat[mt*16*CDIM + i];
  __syncthreads();
  int lane = t & 31, wave = t >> 5;
  int m = lane & 15, kb8 = (lane >> 4) * 16;
  int col = lane & 15, kb16 = (lane >> 4) * 16, ro = (lane >> 4) * 8;
  v16h A = fragA(&at[m*CDIM + kb8]);
  for (int nt = wave; nt < 96; nt += 8) {
    int n0 = nt * 16;
    v16h B;
    { const float* bp = head_w + (size_t)(n0 + col) * CDIM + kb16;
      #pragma unroll
      for (int k = 0; k < 16; ++k) B[k] = (h16)bp[k]; }
    v8f acc = {0,0,0,0,0,0,0,0};
    acc = wmma32(A, B, acc);
    #pragma unroll
    for (int j = 0; j < 8; ++j) {
      int row = mt*16 + ro + j;
      p[(size_t)row*1536 + n0 + col] = acc[j] + head_b[n0 + col];
    }
  }
}

// ---------------------------------------------------------------- head nonlinearities
__global__ void k_params(const float* __restrict__ p, const float* __restrict__ dt,
                         float* __restrict__ Are, float* __restrict__ Aim,
                         float* __restrict__ SG)
{
  int idx = blockIdx.x*256 + threadIdx.x;          // ((l*C+c)*WF+j)*R+r
  if (idx >= LDIM*CDIM*WF*RDIM) return;
  int r = idx & 15;
  int j = (idx >> 4) % WF;
  int c = (idx / (RDIM*WF)) & 31;
  int l = idx / (RDIM*WF*CDIM);
  const float* pp = p + (size_t)(l*WF + j)*1536 + (c*RDIM + r)*3;
  float p0 = pp[0], p1 = pp[1], p2 = pp[2];
  float nu = (p0 > 20.f) ? p0 : log1pf(__expf(p0));
  float th = tanhf(p1) * PI_F;
  float d  = dt[l];
  float dec = __expf(-nu * d);
  float sn, cs; __sincosf(th * d, &sn, &cs);
  Are[idx] = dec * cs; Aim[idx] = dec * sn; SG[idx] = sigmf(p2);
}

// ---------------------------------------------------------------- rfft (direct DFT)
__global__ void k_fft(const float* __restrict__ x, float* __restrict__ xfre,
                      float* __restrict__ xfim)
{
  __shared__ float xs[HDIM][WDIM];
  __shared__ float ct[WF][WDIM], st[WF][WDIM];
  int l = blockIdx.x & 15, c = blockIdx.x >> 4;
  int t = threadIdx.x;
  for (int i = t; i < WF*WDIM; i += 256) {
    int f = i >> 6, w = i & 63;
    float s, co; __sincosf(-2.f * PI_F * (float)(f * w) / 64.f, &s, &co);
    ct[f][w] = co; st[f][w] = s;
  }
  for (int i = t; i < HDIM*WDIM; i += 256) {
    int h = i >> 6, w = i & 63;
    xs[h][w] = x[((size_t)(c*LDIM + l)*HDIM + h)*WDIM + w];
  }
  __syncthreads();
  for (int i = t; i < HDIM*WF; i += 256) {
    int h = i / WF, f = i % WF;
    float re = 0.f, im = 0.f;
    for (int w = 0; w < WDIM; ++w) { re += xs[h][w]*ct[f][w]; im += xs[h][w]*st[f][w]; }
    size_t o = (((size_t)(l*CDIM + c)*HDIM + h)*WF) + f;
    xfre[o] = re; xfim[o] = im;
  }
}

// ---------------------------------------------------------------- recurrence scan
__global__ void k_scan(const float* __restrict__ Are, const float* __restrict__ Aim,
                       const float* __restrict__ SG,  const float* __restrict__ xfre,
                       const float* __restrict__ xfim, const float* __restrict__ dt,
                       h16* __restrict__ hbuf)
{
  int idx = blockIdx.x*256 + threadIdx.x;
  if (idx >= CDIM*HDIM*WF*RDIM) return;
  int r = idx & 15;
  int f = (idx >> 4) % WF;
  int h = (idx / (16*WF)) & 63;
  int c = idx / (16*WF*HDIM);
  float hre = 0.f, him = 0.f;
  for (int l = 0; l < LDIM; ++l) {
    size_t pa = ((size_t)((l*CDIM + c)*WF + f))*RDIM + r;
    size_t px = (((size_t)(l*CDIM + c)*HDIM + h)*WF) + f;
    float are = Are[pa], aim = Aim[pa], sg = SG[pa];
    float g = 1.f - __expf(-2.f * fmaxf(dt[l], 0.f));
    float ure = sg * g * xfre[px], uim = sg * g * xfim[px];
    float nre = are*hre - aim*him + ure;
    float nim = are*him + aim*hre + uim;
    hre = nre; him = nim;
    hbuf[px*32 + r]      = (h16)hre;   // row=(l,c,h,f): cols 0..15=re, 16..31=im
    hbuf[px*32 + 16 + r] = (h16)him;
  }
}

// ---------------------------------------------------------------- mix GEMM (WMMA + TDM)
__global__ void k_mix(const h16* __restrict__ hbuf, const float* __restrict__ mix_w,
                      const float* __restrict__ mix_b, const float* __restrict__ rank_scale,
                      const float* __restrict__ proj_w, const float* __restrict__ proj_b,
                      float* __restrict__ yfre, float* __restrict__ yfim)
{
  alignas(16) __shared__ h16 bw[32*32];
#if HAVE_TDM
  alignas(16) __shared__ h16 atile[8][16*CDIM];     // per-wave A tile (1KB each)
#endif
  __shared__ float sr[16], mb[32];
  int t = threadIdx.x;
  int lane = t & 31, wave = t >> 5;
  size_t row0 = ((size_t)blockIdx.x * 8 + wave) * 16;

#if HAVE_TDM
  // Kick the TDM DMA for this wave's A tile before touching B.
  tdm_load_1d(hbuf + row0*32, &atile[wave][0], 16*CDIM*2);
#endif
  for (int i = t; i < 1024; i += 256) bw[i] = (h16)mix_w[i];   // Bt[n][k]=mix_w[n][k]
  if (t < 16) sr[t] = rank_scale[t] * proj_w[t];
  if (t < 32) mb[t] = mix_b[t];
  __syncthreads();

  int m = lane & 15, kb8 = (lane >> 4) * 16;
  int col = lane & 15, kb16 = (lane >> 4) * 16;
  __builtin_prefetch(hbuf + (row0 + 128) * 32, 0, 1);          // global_prefetch_b8

  v16h B0 = fragBt(&bw[col*32 + kb16]);
  v16h B1 = fragBt(&bw[(16 + col)*32 + kb16]);
#if HAVE_TDM
  tdm_wait();                                                  // s_wait_tensorcnt 0
  v16h A = fragA(&atile[wave][m*32 + kb8]);
#else
  v16h A = fragA(hbuf + (row0 + m)*32 + kb8);
#endif
  v8f c0 = {0,0,0,0,0,0,0,0}, c1 = {0,0,0,0,0,0,0,0};
  c0 = wmma32(A, B0, c0);
  c1 = wmma32(A, B1, c1);

  float s = sr[col], b0 = mb[col], b1 = mb[16 + col], pb = proj_b[0];
  float re[8], im[8];
  #pragma unroll
  for (int j = 0; j < 8; ++j) { re[j] = (c0[j] + b0) * s; im[j] = (c1[j] + b1) * s; }
  #pragma unroll
  for (int mask = 1; mask < 16; mask <<= 1) {
    #pragma unroll
    for (int j = 0; j < 8; ++j) {
      re[j] += __shfl_xor(re[j], mask, 32);
      im[j] += __shfl_xor(im[j], mask, 32);
    }
  }
  if (col == 0) {
    int ro = (lane >> 4) * 8;
    #pragma unroll
    for (int j = 0; j < 8; ++j) {
      size_t row = row0 + ro + j;
      yfre[row] = re[j] + pb;
      yfim[row] = im[j];
    }
  }
}

// ---------------------------------------------------------------- irfft (direct)
__global__ void k_ifft(const float* __restrict__ yfre, const float* __restrict__ yfim,
                       h16* __restrict__ yspat)
{
  __shared__ float yre[HDIM][WF], yim[HDIM][WF];
  int l = blockIdx.x & 15, c = blockIdx.x >> 4;
  int t = threadIdx.x;
  for (int i = t; i < HDIM*WF; i += 256) {
    int h = i / WF, f = i % WF;
    size_t o = (((size_t)(l*CDIM + c)*HDIM + h)*WF) + f;
    yre[h][f] = yfre[o]; yim[h][f] = yfim[o];
  }
  __syncthreads();
  for (int i = t; i < HDIM*WDIM; i += 256) {
    int h = i >> 6, w = i & 63;
    float acc = yre[h][0] + yre[h][32] * ((w & 1) ? -1.f : 1.f);
    for (int f = 1; f < 32; ++f) {
      float s, co; __sincosf(2.f * PI_F * (float)(f * w) / 64.f, &s, &co);
      acc += 2.f * (yre[h][f]*co - yim[h][f]*s);
    }
    yspat[((size_t)((l*HDIM + h)*WDIM + w) << 5) + c] = (h16)(acc * (1.f/64.f));
  }
}

// ---------------------------------------------------------------- pif GEMM (WMMA + TDM) + GN partials
__global__ void k_pif(const h16* __restrict__ yspat, const float* __restrict__ pif_w,
                      const float* __restrict__ pif_b, float* __restrict__ pifo,
                      float* __restrict__ stats)
{
  alignas(16) __shared__ h16 bw[1024];
#if HAVE_TDM
  alignas(16) __shared__ h16 atile[8][16*CDIM];
#endif
  __shared__ float bsh[32];
  __shared__ float gsum[4], gsq[4];
  int t = threadIdx.x;
  int lane = t & 31, wave = t >> 5;
  int l = blockIdx.x >> 5;
  int tt = (blockIdx.x & 31) * 8 + wave;                       // 256 tiles per frame
  size_t row0 = (size_t)l * NPIX + (size_t)tt * 16;

#if HAVE_TDM
  tdm_load_1d(yspat + row0*32, &atile[wave][0], 16*CDIM*2);
#endif
  for (int i = t; i < 1024; i += 256) bw[i] = (h16)pif_w[i];   // Bt[o][c]=pif_w[o][c]
  if (t < 32) bsh[t] = pif_b[t];
  if (t < 4) { gsum[t] = 0.f; gsq[t] = 0.f; }
  __syncthreads();

  int m = lane & 15, kb8 = (lane >> 4) * 16;
  int col = lane & 15, kb16 = (lane >> 4) * 16, ro = (lane >> 4) * 8;

  v16h B0 = fragBt(&bw[col*32 + kb16]);
  v16h B1 = fragBt(&bw[(16 + col)*32 + kb16]);
#if HAVE_TDM
  tdm_wait();
  v16h A = fragA(&atile[wave][m*32 + kb8]);
#else
  v16h A = fragA(yspat + (row0 + m)*32 + kb8);
#endif
  v8f c0 = {0,0,0,0,0,0,0,0}, c1 = {0,0,0,0,0,0,0,0};
  c0 = wmma32(A, B0, c0);
  c1 = wmma32(A, B1, c1);

  float s0 = 0.f, q0 = 0.f, s1 = 0.f, q1 = 0.f;
  #pragma unroll
  for (int j = 0; j < 8; ++j) {
    size_t row = row0 + ro + j;
    float v0 = c0[j] + bsh[col];
    float v1 = c1[j] + bsh[16 + col];
    pifo[row*32 + col]      = v0;
    pifo[row*32 + 16 + col] = v1;
    s0 += v0; q0 += v0*v0; s1 += v1; q1 += v1*v1;
  }
  atomicAdd(&gsum[col >> 3], s0);       atomicAdd(&gsq[col >> 3], q0);
  atomicAdd(&gsum[2 + (col >> 3)], s1); atomicAdd(&gsq[2 + (col >> 3)], q1);
  __syncthreads();
  if (t < 4) {
    atomicAdd(&stats[l*8 + t*2],     gsum[t]);
    atomicAdd(&stats[l*8 + t*2 + 1], gsq[t]);
  }
}

__global__ void k_zero(float* __restrict__ p, int n) {
  int i = blockIdx.x*256 + threadIdx.x; if (i < n) p[i] = 0.f;
}

__global__ void k_stats(const float* __restrict__ stats, float* __restrict__ musig)
{
  int i = threadIdx.x;                 // 64 = L*GROUPS
  if (i < 64) {
    float n = 8.f * 4096.f;
    float mu = stats[i*2] / n;
    float var = stats[i*2 + 1] / n - mu*mu;
    musig[i*2] = mu; musig[i*2 + 1] = rsqrtf(var + 1e-5f);
  }
}

// ---------------------------------------------------------------- gate MLP (per l,h)
__global__ void k_gate(const float* __restrict__ pifo, const float* __restrict__ musig,
                       const float* __restrict__ gn_w, const float* __restrict__ gn_b,
                       const float* __restrict__ g1w, const float* __restrict__ g1b,
                       const float* __restrict__ g2w, const float* __restrict__ g2b,
                       float* __restrict__ gate)
{
  __shared__ float zb[32]; __shared__ float am[4];
  int l = blockIdx.x >> 6, h = blockIdx.x & 63;
  int c = threadIdx.x;
  size_t base = ((size_t)(l*HDIM + h)*WDIM) * 32 + c;
  float s = 0.f;
  for (int w = 0; w < WDIM; ++w) s += pifo[base + (size_t)w*32];
  float yb = s * (1.f / WDIM);
  int g = c >> 3;
  float mu = musig[(l*4 + g)*2], rs = musig[(l*4 + g)*2 + 1];
  zb[c] = (yb - mu) * rs * gn_w[c] + gn_b[c];
  __syncthreads();
  if (c < 4) {
    float a = g1b[c];
    for (int k = 0; k < 32; ++k) a += zb[k] * g1w[c*32 + k];
    am[c] = siluf(a);
  }
  __syncthreads();
  float o = g2b[c];
  for (int k = 0; k < 4; ++k) o += am[k] * g2w[c*4 + k];
  gate[(size_t)(l*HDIM + h)*32 + c] = sigmf(o);
}

// ---------------------------------------------------------------- final residual
__global__ void k_final(const float* __restrict__ x, const float* __restrict__ pifo,
                        const float* __restrict__ musig, const float* __restrict__ gn_w,
                        const float* __restrict__ gn_b, const float* __restrict__ gate,
                        float* __restrict__ out)
{
  size_t idx = (size_t)blockIdx.x*256 + threadIdx.x;           // (c,l,h,w) flat
  if (idx >= (size_t)CDIM*LDIM*HDIM*WDIM) return;
  int w = idx & 63, h = (idx >> 6) & 63, l = (idx >> 12) & 15, c = (int)(idx >> 16);
  float v = pifo[(((size_t)(l*HDIM + h)*WDIM) + w)*32 + c];
  int g = c >> 3;
  float mu = musig[(l*4 + g)*2], rs = musig[(l*4 + g)*2 + 1];
  float z = (v - mu) * rs * gn_w[c] + gn_b[c];
  out[idx] = x[idx] + z * gate[(size_t)(l*HDIM + h)*32 + c];
}

// ---------------------------------------------------------------- launch
extern "C" void kernel_launch(void* const* d_in, const int* in_sizes, int n_in,
                              void* d_out, int out_size, void* d_ws, size_t ws_size,
                              hipStream_t stream) {
  (void)in_sizes; (void)n_in; (void)out_size; (void)ws_size;
  const float* x       = (const float*)d_in[0];
  const float* dt      = (const float*)d_in[1];
  const float* conv1_w = (const float*)d_in[2];
  const float* conv1_b = (const float*)d_in[3];
  const float* conv2_w = (const float*)d_in[4];
  const float* conv2_b = (const float*)d_in[5];
  const float* gfc_w   = (const float*)d_in[6];
  const float* gfc_b   = (const float*)d_in[7];
  const float* head_w  = (const float*)d_in[8];
  const float* head_b  = (const float*)d_in[9];
  const float* mix_w   = (const float*)d_in[10];
  const float* mix_b   = (const float*)d_in[11];
  const float* rank_sc = (const float*)d_in[12];
  const float* proj_w  = (const float*)d_in[13];
  const float* proj_b  = (const float*)d_in[14];
  const float* pif_w   = (const float*)d_in[15];
  const float* pif_b   = (const float*)d_in[16];
  const float* gn_w    = (const float*)d_in[17];
  const float* gn_b    = (const float*)d_in[18];
  const float* g1w     = (const float*)d_in[19];
  const float* g1b     = (const float*)d_in[20];
  const float* g2w     = (const float*)d_in[21];
  const float* g2b     = (const float*)d_in[22];
  float* out = (float*)d_out;

  char* ws = (char*)d_ws;
  size_t off = 0;
  auto alloc = [&](size_t bytes) -> void* {
    void* p = ws + off; off = (off + bytes + 255) & ~(size_t)255; return p;
  };
  const size_t NROW = (size_t)LDIM*CDIM*HDIM*WF;        // 1,081,344
  h16*   f1    = (h16*)  alloc((size_t)LDIM*NPIX*CDIM*2);
  h16*   f2    = (h16*)  alloc((size_t)LDIM*NPIX*CDIM*2);
  float* glob  = (float*)alloc(LDIM*CDIM*4);
  h16*   feat  = (h16*)  alloc((size_t)LDIM*WF*CDIM*2);
  float* phead = (float*)alloc((size_t)LDIM*WF*1536*4);
  float* Are   = (float*)alloc((size_t)LDIM*CDIM*WF*RDIM*4);
  float* Aim   = (float*)alloc((size_t)LDIM*CDIM*WF*RDIM*4);
  float* SG    = (float*)alloc((size_t)LDIM*CDIM*WF*RDIM*4);
  float* xfre  = (float*)alloc(NROW*4);
  float* xfim  = (float*)alloc(NROW*4);
  h16*   hbuf  = (h16*)  alloc(NROW*32*2);
  float* yfre  = (float*)alloc(NROW*4);
  float* yfim  = (float*)alloc(NROW*4);
  h16*   yspat = (h16*)  alloc((size_t)LDIM*NPIX*CDIM*2);
  float* pifo  = (float*)alloc((size_t)LDIM*NPIX*CDIM*4);
  float* stats = (float*)alloc(128*4);
  float* musig = (float*)alloc(128*4);
  float* gate  = (float*)alloc((size_t)LDIM*HDIM*CDIM*4);

  k_conv<true ><<<LDIM*8, 256, 0, stream>>>(x,  nullptr, conv1_w, conv1_b, f1);
  k_conv<false><<<LDIM*8, 256, 0, stream>>>(nullptr, f1, conv2_w, conv2_b, f2);
  k_glob<<<LDIM, 256, 0, stream>>>(f2, gfc_w, gfc_b, glob);
  k_pool<<<LDIM, 256, 0, stream>>>(f2, glob, feat);
  k_head<<<33, 256, 0, stream>>>(feat, head_w, head_b, phead);
  k_params<<<(LDIM*CDIM*WF*RDIM + 255)/256, 256, 0, stream>>>(phead, dt, Are, Aim, SG);
  k_fft<<<LDIM*CDIM, 256, 0, stream>>>(x, xfre, xfim);
  k_scan<<<(CDIM*HDIM*WF*RDIM + 255)/256, 256, 0, stream>>>(Are, Aim, SG, xfre, xfim, dt, hbuf);
  k_mix<<<(int)(NROW/16/8), 256, 0, stream>>>(hbuf, mix_w, mix_b, rank_sc, proj_w, proj_b,
                                              yfre, yfim);
  k_ifft<<<LDIM*CDIM, 256, 0, stream>>>(yfre, yfim, yspat);
  k_zero<<<1, 256, 0, stream>>>(stats, 128);
  k_pif<<<LDIM*32, 256, 0, stream>>>(yspat, pif_w, pif_b, pifo, stats);
  k_stats<<<1, 64, 0, stream>>>(stats, musig);
  k_gate<<<LDIM*HDIM, 32, 0, stream>>>(pifo, musig, gn_w, gn_b, g1w, g1b, g2w, g2b, gate);
  k_final<<<(int)(((size_t)CDIM*LDIM*HDIM*WDIM + 255)/256), 256, 0, stream>>>(
      x, pifo, musig, gn_w, gn_b, gate, out);
}
